// LogoAwareAttention_38740605010378
// MI455X (gfx1250) — matrix operand order repaired and checked by reading, not compiled, not run on hardware
//
#include <hip/hip_runtime.h>
#include <hip/hip_bf16.h>
#include <cstdint>
#include <cstddef>

// ---------------- problem constants ----------------
#define B_  8
#define N_  1024
#define C_  768
#define H_  12
#define DH  64
#define M_TOT (B_ * N_)      // 8192
#define QKV_N (3 * C_)       // 2304

typedef __attribute__((ext_vector_type(16))) __bf16 v16bf;
typedef __attribute__((ext_vector_type(2)))  __bf16 v2bf;
typedef __attribute__((ext_vector_type(8)))  float  v8f;
typedef __attribute__((ext_vector_type(4)))  uint32_t u32x4;
typedef __attribute__((ext_vector_type(8)))  uint32_t u32x8;

// float -> bf16 bits, round-to-nearest-even (scalar)
__device__ __forceinline__ unsigned short f2bf(float f) {
    union { float f; uint32_t u; } c; c.f = f;
    uint32_t u = c.u;
    uint32_t r = u + 0x7FFFu + ((u >> 16) & 1u);
    return (unsigned short)(r >> 16);
}

// two floats -> packed bf16x2 dword
__device__ __forceinline__ uint32_t pk2bf(float lo, float hi) {
#if __has_builtin(__builtin_amdgcn_cvt_pk_bf16_f32)
    union { v2bf v; uint32_t u; } c;
    c.v = __builtin_amdgcn_cvt_pk_bf16_f32(lo, hi);
    return c.u;
#else
    return (uint32_t)f2bf(lo) | ((uint32_t)f2bf(hi) << 16);
#endif
}

// LDS byte offset of a __shared__ object (flat LDS aperture: addr[31:0])
__device__ __forceinline__ uint32_t lds_off(const void* p) {
    return (uint32_t)(uintptr_t)p;
}

// async copy 16B global -> LDS (ASYNCcnt-tracked, no VGPR data path)
__device__ __forceinline__ void async_b128(uint32_t lds, const void* gptr) {
    asm volatile("global_load_async_to_lds_b128 %0, %1, off"
                 :: "v"(lds), "v"((uint64_t)(uintptr_t)gptr) : "memory");
}
__device__ __forceinline__ void wait_async0() {
    asm volatile("s_wait_asynccnt 0x0" ::: "memory");
}
__device__ __forceinline__ void wait_ds0() {
    asm volatile("s_wait_dscnt 0x0" ::: "memory");
}

// -------- Tensor Data Mover: one DMA descriptor moves a whole tile --------
// Issue a TDM load with D# group0 (4 SGPRs) + group1 (8 SGPRs); 2D max.
__device__ __forceinline__ void tdm_load(u32x4 g0, u32x8 g1) {
    asm volatile("tensor_load_to_lds %0, %1" :: "s"(g0), "s"(g1) : "memory");
}

// Build D# groups for a 2D tile of 8-byte elements and issue the DMA.
//  rows x row_units tile; global row stride `stride_units` (8B units).
__device__ __forceinline__ void tdm_load_tile(
    uint32_t lds, const void* gptr,
    uint32_t row_units, uint32_t rows, uint32_t stride_units)
{
    const uint64_t ga = (uint64_t)(uintptr_t)gptr;
    u32x4 g0;
    g0[0] = 1u;                                    // count=1, user mode
    g0[1] = lds;                                   // lds_addr (bytes)
    g0[2] = (uint32_t)ga;                          // global_addr[31:0]
    g0[3] = (uint32_t)(ga >> 32) | (2u << 30);     // global_addr[56:32], type=2
    u32x8 g1;
    g1[0] = 3u << 16;                              // data_size=3 (8B)
    g1[1] = (stride_units & 0xFFFFu) << 16;        // tensor_dim0[15:0]
    g1[2] = (stride_units >> 16) | (rows << 16);   // tensor_dim0[31:16] | tensor_dim1[15:0]
    g1[3] = row_units << 16;                       // tensor_dim1[31:16]=0 | tile_dim0
    g1[4] = rows;                                  // tile_dim1 | tile_dim2=0
    g1[5] = stride_units;                          // tensor_dim0_stride[31:0]
    g1[6] = 0u;                                    // stride hi | tensor_dim1_stride lo
    g1[7] = 0u;
    tdm_load(g0, g1);
}

// 16x(K=32) bf16 WMMA fragment: 16 bf16 per lane = two 16-byte chunks
union Frag {
    v16bf v;
    uint4 q[2];
};

__device__ __forceinline__ v8f wmma_bf16(const Frag& a, const Frag& b, v8f c) {
    return __builtin_amdgcn_wmma_f32_16x16x32_bf16(
        false, a.v, false, b.v, (short)0, c, false, false);
}

// =====================================================================
// Prep 1: f32 -> bf16 elementwise (x). One float4 -> one bf16x4 per thread.
// =====================================================================
__global__ __launch_bounds__(256) void cvt_bf16_kernel(
    const float* __restrict__ src, unsigned short* __restrict__ dst)
{
    const size_t i = (size_t)blockIdx.x * 256 + threadIdx.x;
    const float4 v = *(const float4*)&src[i * 4];
    uint32_t* d = (uint32_t*)&dst[i * 4];
    d[0] = pk2bf(v.x, v.y);
    d[1] = pk2bf(v.z, v.w);
}

// =====================================================================
// Prep 2: transpose + convert weights: W[K][Nc] f32 -> WT[Nc][K] bf16.
// 32x32 LDS tile; paid once so the GEMM hot loops are pure copies.
// =====================================================================
__global__ __launch_bounds__(128) void transpose_cvt_kernel(
    const float* __restrict__ W, unsigned short* __restrict__ WT,
    int K, int Nc)
{
    __shared__ __align__(16) unsigned short lT[32 * 32];   // [nn][kk]
    const int k0 = blockIdx.x * 32;
    const int n0 = blockIdx.y * 32;

    #pragma unroll
    for (int i = 0; i < 2; ++i) {
        int u  = threadIdx.x + i * 128;     // 256 float4
        int kk = u >> 3, c4 = u & 7;
        const float4 v = *(const float4*)&W[(size_t)(k0 + kk) * Nc + n0 + c4 * 4];
        lT[(c4 * 4 + 0) * 32 + kk] = f2bf(v.x);
        lT[(c4 * 4 + 1) * 32 + kk] = f2bf(v.y);
        lT[(c4 * 4 + 2) * 32 + kk] = f2bf(v.z);
        lT[(c4 * 4 + 3) * 32 + kk] = f2bf(v.w);
    }
    __syncthreads();
    const int nn = threadIdx.x >> 2, c = threadIdx.x & 3;   // 128 uint4
    *(uint4*)&WT[(size_t)(n0 + nn) * K + k0 + c * 8] = ((const uint4*)&lT[nn * 32])[c];
}

// =====================================================================
// Generic bf16 GEMM core: 64x64 tile / workgroup, 128 threads (4 waves),
// K-step 32, double-buffered LDS tiles filled entirely with
// global_load_async_to_lds_b128; inner loop = ds_load_b128 + v_wmma.
//   A: [M][K] bf16 row-major, B: BT[N][K] bf16 (pre-transposed).
// =====================================================================
template <int KDIM>
__device__ __forceinline__ void gemm_tile_64x64(
    const unsigned short* __restrict__ A, const unsigned short* __restrict__ BT,
    int m0, int n0, int tid, int wave, int hi, int lr,
    unsigned short (&lA)[2][64 * 32], unsigned short (&lB)[2][64 * 32],
    v8f (&acc)[4])
{
    auto issue = [&](int kt, int buf) {
        #pragma unroll
        for (int i = 0; i < 2; ++i) {
            int u = tid + i * 128, row = u >> 2, c = u & 3;   // 256 x 16B
            async_b128(lds_off(&lA[buf][row * 32 + c * 8]),
                       &A[(size_t)(m0 + row) * KDIM + kt * 32 + c * 8]);
        }
        #pragma unroll
        for (int i = 0; i < 2; ++i) {
            int u = tid + i * 128, nn = u >> 2, c = u & 3;
            async_b128(lds_off(&lB[buf][nn * 32 + c * 8]),
                       &BT[(size_t)(n0 + nn) * KDIM + kt * 32 + c * 8]);
        }
    };

    issue(0, 0);
    for (int kt = 0; kt < KDIM / 32; ++kt) {
        const int cur = kt & 1;
        wait_async0();                       // tiles for kt are in LDS
        __syncthreads();
        if (kt + 1 < KDIM / 32) issue(kt + 1, 1 - cur);   // stream next tile

        Frag a;
        const uint4* pa = (const uint4*)&lA[cur][(wave * 16 + lr) * 32];
        a.q[0] = pa[hi];
        a.q[1] = pa[2 + hi];
        #pragma unroll
        for (int ct = 0; ct < 4; ++ct) {
            Frag b;
            const uint4* pb = (const uint4*)&lB[cur][(ct * 16 + lr) * 32];
            b.q[0] = pb[hi];
            b.q[1] = pb[2 + hi];
            acc[ct] = wmma_bf16(a, b, acc[ct]);
        }
        wait_ds0();                          // reads of buffer `cur` done
        __syncthreads();
    }
}

// =====================================================================
// Kernel 1: QKV GEMM xb[8192x768](bf16) x WqkvT[2304x768](bf16);
// C scattered as bf16 into q[b,h,n,d], k[b,h,n,d], vT[b,h,d,n].
// =====================================================================
__global__ __launch_bounds__(128) void qkv_gemm_kernel(
    const unsigned short* __restrict__ xb, const unsigned short* __restrict__ WT,
    unsigned short* __restrict__ q, unsigned short* __restrict__ k,
    unsigned short* __restrict__ vT)
{
    __shared__ __align__(16) unsigned short lA[2][64 * 32];
    __shared__ __align__(16) unsigned short lB[2][64 * 32];

    const int tid  = threadIdx.x;
    const int wave = tid >> 5;
    const int lane = tid & 31;
    const int hi   = (lane >= 16) ? 1 : 0;
    const int lr   = lane & 15;
    const int m0   = blockIdx.x * 64;
    const int n0t  = blockIdx.y * 64;

    v8f acc[4] = {};
    gemm_tile_64x64<C_>(xb, WT, m0, n0t, tid, wave, hi, lr, lA, lB, acc);

    const int which = n0t / C_;
    const int rem   = n0t - which * C_;
    const int h     = rem / DH;
    #pragma unroll
    for (int ct = 0; ct < 4; ++ct) {
        const int d = ct * 16 + lr;
        #pragma unroll
        for (int r = 0; r < 8; ++r) {
            const int grow = m0 + wave * 16 + r + hi * 8;
            const int b    = grow >> 10;
            const int n    = grow & (N_ - 1);
            const unsigned short val = f2bf(acc[ct][r]);
            if (which == 0)      q[((size_t)(b * H_ + h) * N_ + n) * DH + d] = val;
            else if (which == 1) k[((size_t)(b * H_ + h) * N_ + n) * DH + d] = val;
            else                 vT[((size_t)(b * H_ + h) * DH + d) * N_ + n] = val;
        }
    }
}

// =====================================================================
// Kernel 2: flash attention per (b,h). Per-head scalar logit bias is
// softmax-shift-invariant -> dropped. Grid (B*H, N/64), 128 threads.
// K/V 64-key blocks DMA'd into double-buffered LDS by the Tensor Data
// Mover (wave0: K descriptor, wave1: V descriptor; TENSORcnt + barrier
// publishes the buffer). Each wave: 16 query rows, online softmax,
// P through per-wave LDS for the PV WMMA.
// =====================================================================
__global__ __launch_bounds__(128) void attention_kernel(
    const unsigned short* __restrict__ q, const unsigned short* __restrict__ k,
    const unsigned short* __restrict__ vT, unsigned short* __restrict__ attn_out)
{
    __shared__ __align__(16) unsigned short lK[2][64 * 64];  // [key][d]
    __shared__ __align__(16) unsigned short lV[2][64 * 64];  // [d][key]
    __shared__ __align__(16) unsigned short lP[4][16 * 64];  // per-wave P

    const int tid  = threadIdx.x;
    const int bh   = blockIdx.x;
    const int b    = bh / H_;
    const int h    = bh - b * H_;
    const int wave = tid >> 5;
    const int lane = tid & 31;
    const int hi   = (lane >= 16) ? 1 : 0;
    const int lr   = lane & 15;

    const unsigned short* qp = q  + (size_t)bh * N_ * DH;
    const unsigned short* kp = k  + (size_t)bh * N_ * DH;
    const unsigned short* vp = vT + (size_t)bh * DH * N_;

    const int qrow0 = blockIdx.y * 64 + wave * 16;
    const float scale = 0.125f;   // Dh^-0.5

    // one TDM descriptor per matrix per block: K = contiguous 8KB
    // (1 x 1024 8B-units), V = 64 rows x 16 units, row stride 256 units.
    auto issue_kv = [&](int kb, int buf) {
        const int kbase = kb * 64;
        if (wave == 0) {
            tdm_load_tile(lds_off(&lK[buf][0]), kp + (size_t)kbase * DH,
                          /*row_units=*/1024u, /*rows=*/1u, /*stride=*/1024u);
        } else if (wave == 1) {
            tdm_load_tile(lds_off(&lV[buf][0]), vp + kbase,
                          /*row_units=*/16u, /*rows=*/64u, /*stride=*/256u);
        }
    };

    Frag qf[2];
    {
        const uint4* pq = (const uint4*)&qp[(size_t)(qrow0 + lr) * DH];
        #pragma unroll
        for (int kc = 0; kc < 2; ++kc) {
            qf[kc].q[0] = pq[kc * 4 + hi];
            qf[kc].q[1] = pq[kc * 4 + 2 + hi];
        }
    }

    v8f  o[4] = {};
    float mrow[8], lrow[8];
    #pragma unroll
    for (int r = 0; r < 8; ++r) { mrow[r] = -INFINITY; lrow[r] = 0.0f; }

    unsigned short* pw = lP[wave];

    issue_kv(0, 0);
    for (int kb = 0; kb < N_ / 64; ++kb) {
        const int cur = kb & 1;
        __builtin_amdgcn_s_wait_tensorcnt(0);   // issuing waves drain their DMA
        __syncthreads();                        // publish buffer `cur` to all
        if (kb + 1 < N_ / 64) issue_kv(kb + 1, 1 - cur);

        // ---- S = scale * Q K^T ----
        v8f s[4] = {};
        #pragma unroll
        for (int ct = 0; ct < 4; ++ct) {
            const uint4* pk = (const uint4*)&lK[cur][(ct * 16 + lr) * 64];
            #pragma unroll
            for (int kc = 0; kc < 2; ++kc) {
                Frag kf;
                kf.q[0] = pk[kc * 4 + hi];
                kf.q[1] = pk[kc * 4 + 2 + hi];
                s[ct] = wmma_bf16(qf[kc], kf, s[ct]);
            }
        }

        // ---- online softmax (row r + hi*8 lives in one 16-lane half) ----
        float cmax[8];
        #pragma unroll
        for (int r = 0; r < 8; ++r) {
            float m = s[0][r] * scale;
            #pragma unroll
            for (int ct = 0; ct < 4; ++ct) { s[ct][r] *= scale; m = fmaxf(m, s[ct][r]); }
            #pragma unroll
            for (int off = 8; off >= 1; off >>= 1)
                m = fmaxf(m, __shfl_xor(m, off, 32));
            cmax[r] = m;
        }
        #pragma unroll
        for (int r = 0; r < 8; ++r) {
            const float mnew  = fmaxf(mrow[r], cmax[r]);
            const float alpha = __expf(mrow[r] - mnew);
            mrow[r] = mnew;
            lrow[r] *= alpha;
            #pragma unroll
            for (int dt = 0; dt < 4; ++dt) o[dt][r] *= alpha;
        }
        float rsum[8] = {};
        #pragma unroll
        for (int ct = 0; ct < 4; ++ct)
            #pragma unroll
            for (int r = 0; r < 8; ++r) {
                const float p = __expf(s[ct][r] - mrow[r]);
                s[ct][r] = p;
                rsum[r] += p;
            }
        #pragma unroll
        for (int r = 0; r < 8; ++r) {
            float t = rsum[r];
            #pragma unroll
            for (int off = 8; off >= 1; off >>= 1) t += __shfl_xor(t, off, 32);
            lrow[r] += t;
        }

        // ---- P -> per-wave LDS bf16 -> A-fragment ----
        #pragma unroll
        for (int ct = 0; ct < 4; ++ct)
            #pragma unroll
            for (int r = 0; r < 8; ++r)
                pw[(r + hi * 8) * 64 + ct * 16 + lr] = f2bf(s[ct][r]);
        wait_ds0();

        Frag pf[2];
        {
            const uint4* pp = (const uint4*)&pw[lr * 64];
            #pragma unroll
            for (int kc = 0; kc < 2; ++kc) {
                pf[kc].q[0] = pp[kc * 4 + hi];
                pf[kc].q[1] = pp[kc * 4 + 2 + hi];
            }
        }

        // ---- O += P V ----
        #pragma unroll
        for (int dt = 0; dt < 4; ++dt) {
            const uint4* pv = (const uint4*)&lV[cur][(dt * 16 + lr) * 64];
            #pragma unroll
            for (int kc = 0; kc < 2; ++kc) {
                Frag vf;
                vf.q[0] = pv[kc * 4 + hi];
                vf.q[1] = pv[kc * 4 + 2 + hi];
                o[dt] = wmma_bf16(pf[kc], vf, o[dt]);
            }
        }
        wait_ds0();
        __syncthreads();
    }

    #pragma unroll
    for (int dt = 0; dt < 4; ++dt)
        #pragma unroll
        for (int r = 0; r < 8; ++r) {
            const int   nrow = qrow0 + r + hi * 8;
            const float val  = o[dt][r] / lrow[r];
            attn_out[((size_t)(b * N_ + nrow)) * C_ + h * DH + dt * 16 + lr] = f2bf(val);
        }
}

// =====================================================================
// Kernel 3: projection GEMM attn[8192x768](bf16) x WprojT[768x768](bf16)
//   + b_proj -> f32 out.
// =====================================================================
__global__ __launch_bounds__(128) void proj_gemm_kernel(
    const unsigned short* __restrict__ attn, const unsigned short* __restrict__ WT,
    const float* __restrict__ bproj, float* __restrict__ out)
{
    __shared__ __align__(16) unsigned short lA[2][64 * 32];
    __shared__ __align__(16) unsigned short lB[2][64 * 32];

    const int tid  = threadIdx.x;
    const int wave = tid >> 5;
    const int lane = tid & 31;
    const int hi   = (lane >= 16) ? 1 : 0;
    const int lr   = lane & 15;
    const int m0   = blockIdx.x * 64;
    const int n0   = blockIdx.y * 64;

    v8f acc[4] = {};
    gemm_tile_64x64<C_>(attn, WT, m0, n0, tid, wave, hi, lr, lA, lB, acc);

    #pragma unroll
    for (int ct = 0; ct < 4; ++ct) {
        const int   col  = n0 + ct * 16 + lr;
        const float bias = bproj[col];
        #pragma unroll
        for (int r = 0; r < 8; ++r) {
            const int grow = m0 + wave * 16 + r + hi * 8;
            out[(size_t)grow * C_ + col] = acc[ct][r] + bias;
        }
    }
}

// =====================================================================
extern "C" void kernel_launch(void* const* d_in, const int* in_sizes, int n_in,
                              void* d_out, int out_size, void* d_ws, size_t ws_size,
                              hipStream_t stream) {
    const float* x     = (const float*)d_in[0];
    // d_in[1..3] (geometric/text/color) and d_in[7..9] (per-head biases) form a
    // per-(b,h) scalar added to every logit -> cancels in softmax. Unused.
    const float* Wqkv  = (const float*)d_in[4];
    const float* Wproj = (const float*)d_in[5];
    const float* bproj = (const float*)d_in[6];
    float* out = (float*)d_out;

    const size_t qkv_elems = (size_t)B_ * H_ * N_ * DH;   // 6.29M bf16 each
    const size_t x_elems   = (size_t)M_TOT * C_;          // 6.29M
    unsigned short* q      = (unsigned short*)d_ws;
    unsigned short* k      = q    + qkv_elems;
    unsigned short* vT     = k    + qkv_elems;
    unsigned short* attn   = vT   + qkv_elems;
    unsigned short* xb     = attn + x_elems;
    unsigned short* WqkvT  = xb   + x_elems;              // [2304][768]
    unsigned short* WprojT = WqkvT + (size_t)QKV_N * C_;  // [768][768]
    (void)in_sizes; (void)n_in; (void)out_size; (void)ws_size;

    // one-time conversions / transposes (bandwidth-trivial at 23.3 TB/s)
    cvt_bf16_kernel<<<(x_elems / 4 + 255) / 256, 256, 0, stream>>>(x, xb);
    transpose_cvt_kernel<<<dim3(C_ / 32, QKV_N / 32), 128, 0, stream>>>(Wqkv, WqkvT, C_, QKV_N);
    transpose_cvt_kernel<<<dim3(C_ / 32, C_ / 32), 128, 0, stream>>>(Wproj, WprojT, C_, C_);

    qkv_gemm_kernel<<<dim3(M_TOT / 64, QKV_N / 64), 128, 0, stream>>>(xb, WqkvT, q, k, vT);
    attention_kernel<<<dim3(B_ * H_, N_ / 64), 128, 0, stream>>>(q, k, vT, attn);
    proj_gemm_kernel<<<dim3(M_TOT / 64, C_ / 64), 128, 0, stream>>>(attn, WprojT, bproj, out);
}
